// MultiHeadAttention_57543971832262
// MI455X (gfx1250) — matrix operand order repaired and checked by reading, not compiled, not run on hardware
//
#include <hip/hip_runtime.h>
#include <hip/hip_bf16.h>
#include <stdint.h>

typedef _Float16 v16h __attribute__((ext_vector_type(16)));
typedef float    v8f  __attribute__((ext_vector_type(8)));

union AFrag {
    v16h     v;
    _Float16 h[16];
    uint32_t u[8];
};

static __device__ __forceinline__ v8f wmma16(const AFrag& a, const AFrag& b, v8f c) {
    return __builtin_amdgcn_wmma_f32_16x16x32_f16(false, a.v, false, b.v, (short)0, c, false, false);
}

// 16 consecutive f16 -> B-fragment lane slice
static __device__ __forceinline__ AFrag load16_f16(const _Float16* p) {
    AFrag r;
    uint4 a = *(const uint4*)(p);
    uint4 b = *(const uint4*)(p + 8);
    r.u[0] = a.x; r.u[1] = a.y; r.u[2] = a.z; r.u[3] = a.w;
    r.u[4] = b.x; r.u[5] = b.y; r.u[6] = b.z; r.u[7] = b.w;
    return r;
}

// A-fragment from f16 memory: 16 consecutive halves at p (contiguous K)
static __device__ __forceinline__ AFrag loadA_f16(const _Float16* p) {
    AFrag r;
    uint4 a = *(const uint4*)(p);
    uint4 b = *(const uint4*)(p + 8);
    r.u[0] = a.x; r.u[1] = a.y; r.u[2] = a.z; r.u[3] = a.w;
    r.u[4] = b.x; r.u[5] = b.y; r.u[6] = b.z; r.u[7] = b.w;
    return r;
}

static __device__ __forceinline__ void cvt4(_Float16* h, float4 f) {
    h[0] = (_Float16)f.x; h[1] = (_Float16)f.y; h[2] = (_Float16)f.z; h[3] = (_Float16)f.w;
}

#define D_MODEL 1024
#define T_SEQ   2048
#define NHEAD   16
#define DK      64

// ---------------------------------------------------------------------------
// f32 -> f16 bulk convert, 8 elements/thread (n must be a multiple of 8)
// ---------------------------------------------------------------------------
__global__ __launch_bounds__(256) void cvt_f32_f16(
    const float* __restrict__ src, _Float16* __restrict__ dst, int n) {
    const int i = (blockIdx.x * 256 + threadIdx.x) * 8;
    if (i < n) {
        union { _Float16 h[8]; uint4 q; } r;
        cvt4(r.h + 0, ((const float4*)(src + i))[0]);
        cvt4(r.h + 4, ((const float4*)(src + i))[1]);
        *(uint4*)(dst + i) = r.q;
    }
}

// ---------------------------------------------------------------------------
// Kernel 1: QKV projection (all-f16 operands, f32 accumulate).
// Y[m,e] = sum_k X16[m,k]*W16[e,k] + bias[e];  M=4096, N=3072, K=1024.
// Wave tile 32(M) x 64(N), double-buffered K loop. grid (16,48), block 256.
// Q,K stored f16 [BH][T][64]; V stored transposed f16 [BH][64][T].
// ---------------------------------------------------------------------------
__global__ __launch_bounds__(256) void qkv_gemm16(
    const _Float16* __restrict__ X, const _Float16* __restrict__ W,
    const float* __restrict__ bias,
    _Float16* __restrict__ Qo, _Float16* __restrict__ Ko, _Float16* __restrict__ Vt) {

    const int lane  = threadIdx.x & 31;
    const int wave  = threadIdx.x >> 5;
    const int half  = lane >> 4;
    const int lan16 = lane & 15;
    const int mBase = blockIdx.x * 256 + wave * 32;
    const int nBase = blockIdx.y * 64;

    const _Float16* pa0 = X + (size_t)(mBase + lan16) * D_MODEL + half * 16;
    const _Float16* pa1 = pa0 + (size_t)16 * D_MODEL;
    const _Float16* pb0 = W + (size_t)(nBase + lan16) * D_MODEL + half * 16;
    const _Float16* pb1 = pb0 + (size_t)16 * D_MODEL;
    const _Float16* pb2 = pb0 + (size_t)32 * D_MODEL;
    const _Float16* pb3 = pb0 + (size_t)48 * D_MODEL;

    v8f acc[4][2];
#pragma unroll
    for (int f = 0; f < 4; ++f) { acc[f][0] = v8f{}; acc[f][1] = v8f{}; }

    AFrag a0 = loadA_f16(pa0), a1 = loadA_f16(pa1);
    AFrag b0 = load16_f16(pb0), b1 = load16_f16(pb1);
    AFrag b2 = load16_f16(pb2), b3 = load16_f16(pb3);

#pragma unroll 2
    for (int k0 = 0; k0 < D_MODEL; k0 += 32) {
        const int kn = k0 + 32;  // final iter overreads 64B into adjacent ws buffer (safe)
        AFrag na0 = loadA_f16(pa0 + kn);
        AFrag na1 = loadA_f16(pa1 + kn);
        AFrag nb0 = load16_f16(pb0 + kn);
        AFrag nb1 = load16_f16(pb1 + kn);
        AFrag nb2 = load16_f16(pb2 + kn);
        AFrag nb3 = load16_f16(pb3 + kn);
        acc[0][0] = wmma16(a0, b0, acc[0][0]);
        acc[0][1] = wmma16(a1, b0, acc[0][1]);
        acc[1][0] = wmma16(a0, b1, acc[1][0]);
        acc[1][1] = wmma16(a1, b1, acc[1][1]);
        acc[2][0] = wmma16(a0, b2, acc[2][0]);
        acc[2][1] = wmma16(a1, b2, acc[2][1]);
        acc[3][0] = wmma16(a0, b3, acc[3][0]);
        acc[3][1] = wmma16(a1, b3, acc[3][1]);
        a0 = na0; a1 = na1; b0 = nb0; b1 = nb1; b2 = nb2; b3 = nb3;
    }

#pragma unroll
    for (int ff = 0; ff < 4; ++ff) {
        const int e      = nBase + ff * 16 + lan16;   // 0..3071
        const float bv   = bias[e];
        const int sec    = e >> 10;                   // 0=Q 1=K 2=V
        const int within = e & 1023;
        const int h      = within >> 6;
        const int d      = within & 63;
#pragma unroll
        for (int r = 0; r < 2; ++r) {
#pragma unroll
            for (int j = 0; j < 8; ++j) {
                const int m  = mBase + r * 16 + j + half * 8;
                const int b  = m >> 11;
                const int t  = m & 2047;
                const int bh = b * NHEAD + h;
                const _Float16 val = (_Float16)(acc[ff][r][j] + bv);
                if (sec == 0)      Qo[((size_t)bh * T_SEQ + t) * DK + d] = val;
                else if (sec == 1) Ko[((size_t)bh * T_SEQ + t) * DK + d] = val;
                else               Vt[((size_t)bh * DK + d) * T_SEQ + t] = val;
            }
        }
    }
}

// ---------------------------------------------------------------------------
// Kernel 2: flash attention.  One wave per 16-row q-tile; online softmax.
// grid (BH=32, T/(16*8)=16), block 256 (8 waves).
// ---------------------------------------------------------------------------
__global__ __launch_bounds__(256) void attn_kernel(
    const _Float16* __restrict__ Q, const _Float16* __restrict__ K,
    const _Float16* __restrict__ Vt, _Float16* __restrict__ ctxOut) {

    __shared__ _Float16 pbuf[8][16 * 32];   // per-wave P staging (C-layout -> A-layout)

    const int lane  = threadIdx.x & 31;
    const int wave  = threadIdx.x >> 5;
    const int half  = lane >> 4;
    const int lan16 = lane & 15;

    const int bh    = blockIdx.x;                    // 0..31
    const int qBase = (blockIdx.y * 8 + wave) * 16;  // q-tile start

    const _Float16* Qb = Q  + (size_t)bh * T_SEQ * DK;
    const _Float16* Kb = K  + (size_t)bh * T_SEQ * DK;
    const _Float16* Vb = Vt + (size_t)bh * DK * T_SEQ;

    // Resident Q A-fragments: 16 rows x 64 d, two 32-wide K-chunks
    AFrag qA[2];
#pragma unroll
    for (int f = 0; f < 2; ++f)
        qA[f] = loadA_f16(Qb + (size_t)(qBase + lan16) * DK + f * 32 + half * 16);

    v8f ctx[4] = {v8f{}, v8f{}, v8f{}, v8f{}};
    float m_run[8], l_run[8];
#pragma unroll
    for (int j = 0; j < 8; ++j) { m_run[j] = -__builtin_inff(); l_run[j] = 0.0f; }

    _Float16* pw = pbuf[wave];
    const float scale = 0.125f;   // 1/sqrt(64)

    for (int kt = 0; kt < T_SEQ; kt += 32) {
        // prefetch next k-tile (speculative; dropped if past end)
        __builtin_prefetch(Kb + (size_t)(kt + 32 + lan16) * DK, 0, 0);
        __builtin_prefetch(Vb + (size_t)lan16 * T_SEQ + kt + 32, 0, 0);

        // ---- scores: S[16 x 32] = Q_tile * K_tile^T ---- (batch all loads first)
        AFrag kb00 = load16_f16(Kb + (size_t)(kt + lan16)      * DK + half * 16);
        AFrag kb01 = load16_f16(Kb + (size_t)(kt + lan16)      * DK + 32 + half * 16);
        AFrag kb10 = load16_f16(Kb + (size_t)(kt + 16 + lan16) * DK + half * 16);
        AFrag kb11 = load16_f16(Kb + (size_t)(kt + 16 + lan16) * DK + 32 + half * 16);
        v8f s0 = v8f{}, s1 = v8f{};
        s0 = wmma16(qA[0], kb00, s0);
        s1 = wmma16(qA[0], kb10, s1);
        s0 = wmma16(qA[1], kb01, s0);
        s1 = wmma16(qA[1], kb11, s1);

        // ---- online softmax (row = j + 8*half; cols across 16-lane half) ----
#pragma unroll
        for (int j = 0; j < 8; ++j) {
            float a0 = s0[j] * scale;
            float a1 = s1[j] * scale;
            float r  = fmaxf(a0, a1);
            r = fmaxf(r, __shfl_xor(r, 8, 32));
            r = fmaxf(r, __shfl_xor(r, 4, 32));
            r = fmaxf(r, __shfl_xor(r, 2, 32));
            r = fmaxf(r, __shfl_xor(r, 1, 32));
            float nm   = fmaxf(m_run[j], r);
            float corr = __expf(m_run[j] - nm);
            m_run[j]   = nm;
            float p0 = __expf(a0 - nm);
            float p1 = __expf(a1 - nm);
            float rs = p0 + p1;
            rs += __shfl_xor(rs, 8, 32);
            rs += __shfl_xor(rs, 4, 32);
            rs += __shfl_xor(rs, 2, 32);
            rs += __shfl_xor(rs, 1, 32);
            l_run[j] = l_run[j] * corr + rs;
#pragma unroll
            for (int ff = 0; ff < 4; ++ff) ctx[ff][j] *= corr;
            const int row = j + half * 8;
            pw[row * 32 + lan16]      = (_Float16)p0;
            pw[row * 32 + 16 + lan16] = (_Float16)p1;
        }

        asm volatile("s_wait_dscnt 0x0" ::: "memory");   // P stores visible to A-layout reads

        // ---- re-read P in A layout: lane row = lan16, contiguous K 16-run by half ----
        AFrag pA;
        {
            const uint32_t* pr  = (const uint32_t*)(pw + lan16 * 32 + half * 16);
            const uint32_t* prh = (const uint32_t*)(pw + lan16 * 32 + half * 16 + 8);
            pA.u[0] = pr[0];  pA.u[1] = pr[1];  pA.u[2] = pr[2];  pA.u[3] = pr[3];
            pA.u[4] = prh[0]; pA.u[5] = prh[1]; pA.u[6] = prh[2]; pA.u[7] = prh[3];
        }

        // ---- ctx += P * V  (V transposed: contiguous along kpos) ----
        AFrag vB0 = load16_f16(Vb + (size_t)(lan16)      * T_SEQ + kt + half * 16);
        AFrag vB1 = load16_f16(Vb + (size_t)(16 + lan16) * T_SEQ + kt + half * 16);
        AFrag vB2 = load16_f16(Vb + (size_t)(32 + lan16) * T_SEQ + kt + half * 16);
        AFrag vB3 = load16_f16(Vb + (size_t)(48 + lan16) * T_SEQ + kt + half * 16);
        ctx[0] = wmma16(pA, vB0, ctx[0]);
        ctx[1] = wmma16(pA, vB1, ctx[1]);
        ctx[2] = wmma16(pA, vB2, ctx[2]);
        ctx[3] = wmma16(pA, vB3, ctx[3]);
    }

    // ---- normalize and store ctx f16 as [B*T, 1024] row-major ----
    const int b = bh >> 4, h = bh & 15;
#pragma unroll
    for (int j = 0; j < 8; ++j) {
        const float inv = 1.0f / l_run[j];
        const int t = qBase + j + half * 8;
        _Float16* po = ctxOut + ((size_t)(b * T_SEQ + t)) * D_MODEL + h * DK + lan16;
        po[0]  = (_Float16)(ctx[0][j] * inv);
        po[16] = (_Float16)(ctx[1][j] * inv);
        po[32] = (_Float16)(ctx[2][j] * inv);
        po[48] = (_Float16)(ctx[3][j] * inv);
    }
}

// ---------------------------------------------------------------------------
// Kernel 3: output projection (all-f16 operands, f32 out).
// out[m,n] = sum_k ctx[m,k]*W16[n,k] + bias[n];  M=4096, N=1024, K=1024.
// Wave tile 32x64, double-buffered. grid (16,16), block 256.
// ---------------------------------------------------------------------------
__global__ __launch_bounds__(256) void out_gemm16(
    const _Float16* __restrict__ Xc, const _Float16* __restrict__ W,
    const float* __restrict__ bias, float* __restrict__ out) {

    const int lane  = threadIdx.x & 31;
    const int wave  = threadIdx.x >> 5;
    const int half  = lane >> 4;
    const int lan16 = lane & 15;
    const int mBase = blockIdx.x * 256 + wave * 32;
    const int nBase = blockIdx.y * 64;

    const _Float16* pa0 = Xc + (size_t)(mBase + lan16) * D_MODEL + half * 16;
    const _Float16* pa1 = pa0 + (size_t)16 * D_MODEL;
    const _Float16* pb0 = W + (size_t)(nBase + lan16) * D_MODEL + half * 16;
    const _Float16* pb1 = pb0 + (size_t)16 * D_MODEL;
    const _Float16* pb2 = pb0 + (size_t)32 * D_MODEL;
    const _Float16* pb3 = pb0 + (size_t)48 * D_MODEL;

    v8f acc[4][2];
#pragma unroll
    for (int f = 0; f < 4; ++f) { acc[f][0] = v8f{}; acc[f][1] = v8f{}; }

    AFrag a0 = loadA_f16(pa0), a1 = loadA_f16(pa1);
    AFrag b0 = load16_f16(pb0), b1 = load16_f16(pb1);
    AFrag b2 = load16_f16(pb2), b3 = load16_f16(pb3);

#pragma unroll 2
    for (int k0 = 0; k0 < D_MODEL; k0 += 32) {
        const int kn = k0 + 32;  // final iter overreads 64B into adjacent ws buffer (safe)
        AFrag na0 = loadA_f16(pa0 + kn);
        AFrag na1 = loadA_f16(pa1 + kn);
        AFrag nb0 = load16_f16(pb0 + kn);
        AFrag nb1 = load16_f16(pb1 + kn);
        AFrag nb2 = load16_f16(pb2 + kn);
        AFrag nb3 = load16_f16(pb3 + kn);
        acc[0][0] = wmma16(a0, b0, acc[0][0]);
        acc[0][1] = wmma16(a1, b0, acc[0][1]);
        acc[1][0] = wmma16(a0, b1, acc[1][0]);
        acc[1][1] = wmma16(a1, b1, acc[1][1]);
        acc[2][0] = wmma16(a0, b2, acc[2][0]);
        acc[2][1] = wmma16(a1, b2, acc[2][1]);
        acc[3][0] = wmma16(a0, b3, acc[3][0]);
        acc[3][1] = wmma16(a1, b3, acc[3][1]);
        a0 = na0; a1 = na1; b0 = nb0; b1 = nb1; b2 = nb2; b3 = nb3;
    }

#pragma unroll
    for (int ff = 0; ff < 4; ++ff) {
        const int n    = nBase + ff * 16 + lan16;
        const float bv = bias[n];
#pragma unroll
        for (int r = 0; r < 2; ++r) {
#pragma unroll
            for (int j = 0; j < 8; ++j) {
                const int m = mBase + r * 16 + j + half * 8;
                out[(size_t)m * D_MODEL + n] = acc[ff][r][j] + bv;
            }
        }
    }
}

// ---------------------------------------------------------------------------
extern "C" void kernel_launch(void* const* d_in, const int* in_sizes, int n_in,
                              void* d_out, int out_size, void* d_ws, size_t ws_size,
                              hipStream_t stream) {
    const float* x     = (const float*)d_in[0];
    const float* w_qkv = (const float*)d_in[1];
    const float* b_qkv = (const float*)d_in[2];
    const float* w_out = (const float*)d_in[3];
    const float* b_out = (const float*)d_in[4];
    float* out = (float*)d_out;

    // Workspace layout (f16 elements).  Order matters: the GEMM pipeline
    // overreads one 32-element K-chunk past Ctx/X16/Wq16/Wo16, so each of
    // those is followed by another ws buffer.  Vt (never overread) is last.
    _Float16* ws = (_Float16*)d_ws;
    const size_t SZ_X  = (size_t)4096 * 1024;   // 4 Mi elems
    const size_t SZ_WQ = (size_t)3072 * 1024;   // 3 Mi
    const size_t SZ_WO = (size_t)1024 * 1024;   // 1 Mi
    _Float16* Ctx = ws;
    _Float16* X16 = Ctx + SZ_X;
    _Float16* Wq  = X16 + SZ_X;
    _Float16* Wo  = Wq + SZ_WQ;
    _Float16* Q   = Wo + SZ_WO;
    _Float16* K   = Q + SZ_X;
    _Float16* Vt  = K + SZ_X;

    // 1) pre-convert f32 operands to f16 (one pass each; resident in L2 after)
    cvt_f32_f16<<<dim3((int)(SZ_X  / 8 / 256)), 256, 0, stream>>>(x,     X16, (int)SZ_X);
    cvt_f32_f16<<<dim3((int)(SZ_WQ / 8 / 256)), 256, 0, stream>>>(w_qkv, Wq,  (int)SZ_WQ);
    cvt_f32_f16<<<dim3((int)(SZ_WO / 8 / 256)), 256, 0, stream>>>(w_out, Wo,  (int)SZ_WO);

    // 2) QKV projection
    qkv_gemm16<<<dim3(16, 48), 256, 0, stream>>>(X16, Wq, b_qkv, Q, K, Vt);
    // 3) attention
    attn_kernel<<<dim3(32, 16), 256, 0, stream>>>(Q, K, Vt, Ctx);
    // 4) output projection
    out_gemm16<<<dim3(16, 16), 256, 0, stream>>>(Ctx, w_out ? Wo : Wo, b_out, out);
}
